// LIFLayer_57535381897772
// MI455X (gfx1250) — matrix operand order repaired and checked by reading, not compiled, not run on hardware
//
#include <hip/hip_runtime.h>

typedef __attribute__((ext_vector_type(16))) _Float16 v16h;
typedef __attribute__((ext_vector_type(8)))  _Float16 v8h;
typedef __attribute__((ext_vector_type(8)))  float    v8f;

// Problem sizes (fixed by reference)
#define B_   128
#define T_   256
#define F_   256
#define K_   256

#define MTILE 64        // rows of spikes per workgroup
// LDS pitches (halfs). 264*2B = 528B rows: 16B-aligned, bank stride 132%64=4
// -> 16 distinct banks across a 16-lane fragment read. 40*2B = 80B rows for
// the W slab: 16B-aligned, bank stride 20 -> 16 distinct banks.
#define LDA  264
#define LDB  40

// ---------------------------------------------------------------------------
// GEMM: currents[M=32768,256] = spikes[M,256] * W[256,256] + bias
// Block: 256 threads (8 wave32s), M-tile 64, N full 256.
// Wave w: m-strip = w&3 (16 rows), n-half = w>>2 (128 cols = 8 x 16-col tiles)
// Full A tile staged once (f32->f16); W^T k-slabs staged per k-iteration.
// ---------------------------------------------------------------------------
__global__ __launch_bounds__(256)
void lif_gemm_wmma(const float* __restrict__ spikes,
                   const float* __restrict__ W,
                   const float* __restrict__ bias,
                   float* __restrict__ currents) {
  __shared__ _Float16 ldsA[MTILE * LDA];  // 33.8 KB : A tile  [64 rows][256 K]
  __shared__ _Float16 ldsB[256 * LDB];    // 20.5 KB : W^T slab[256 N ][32 K]

  const int tid    = threadIdx.x;
  const int wave   = tid >> 5;
  const int lane   = tid & 31;
  const int m0     = blockIdx.x * MTILE;
  const int mstrip = wave & 3;            // which 16-row strip
  const int nbase  = (wave >> 2) * 128;   // which 128-col half

  // ---- stage full A tile once: 64x256 f32 -> f16, fully coalesced ----
  #pragma unroll
  for (int pass = 0; pass < 16; ++pass) {
    const int idx = pass * 1024 + tid * 4;     // linear float index in tile
    const int r   = idx >> 8;                  // row 0..63
    const int c   = idx & 255;                 // col 0..252 (mult of 4)
    const float4 v = *reinterpret_cast<const float4*>(
        &spikes[(size_t)(m0 + r) * K_ + c]);
    _Float16* dst = &ldsA[r * LDA + c];
    dst[0] = (_Float16)v.x; dst[1] = (_Float16)v.y;
    dst[2] = (_Float16)v.z; dst[3] = (_Float16)v.w;
    // Pull the next block's A rows toward L2 (global_prefetch_b8).
    __builtin_prefetch(&spikes[(size_t)(m0 + MTILE + r) * K_ + c], 0, 1);
  }

  v8f acc[8] = {};  // 8 x (16x16 f32) accumulators -> 16x128 per wave

  for (int k0 = 0; k0 < K_; k0 += 32) {
    // ---- stage W^T slab: ldsB[n][k] = W[k0+k][n] (coalesced across tid) ----
    __syncthreads();  // waves done reading previous slab
    #pragma unroll 8
    for (int k = 0; k < 32; ++k) {
      ldsB[tid * LDB + k] = (_Float16)W[(size_t)(k0 + k) * F_ + tid];
    }
    __syncthreads();

    // ---- A fragment (16x32 f16), ISA layout:
    // lane<16: row=lane,     K = {0..7, 16..23}
    // lane>=16: row=lane-16, K = {8..15, 24..31}
    const int arow  = mstrip * 16 + (lane & 15);
    const int khalf = lane >> 4;
    v16h afrag;
    {
      const v8h a0 = *reinterpret_cast<const v8h*>(&ldsA[arow * LDA + k0 + khalf * 8]);
      const v8h a1 = *reinterpret_cast<const v8h*>(&ldsA[arow * LDA + k0 + 16 + khalf * 8]);
      #pragma unroll
      for (int i = 0; i < 8; ++i) { afrag[i] = a0[i]; afrag[i + 8] = a1[i]; }
    }

    // ---- 8 B fragments + WMMA ----
    #pragma unroll
    for (int nt = 0; nt < 8; ++nt) {
      // B 32x16: lanes 0-15 hold K={0..7,16..23} of col N=lane,
      //          lanes 16-31 hold K={8..15,24..31} of col N=lane-16.
      const int ncol = nbase + nt * 16 + (lane & 15);
      const int kg   = lane >> 4;
      v16h bfrag;
      {
        const v8h b0 = *reinterpret_cast<const v8h*>(&ldsB[ncol * LDB + kg * 8]);
        const v8h b1 = *reinterpret_cast<const v8h*>(&ldsB[ncol * LDB + 16 + kg * 8]);
        #pragma unroll
        for (int i = 0; i < 8; ++i) { bfrag[i] = b0[i]; bfrag[i + 8] = b1[i]; }
      }
      acc[nt] = __builtin_amdgcn_wmma_f32_16x16x32_f16(
          /*neg_a=*/false, afrag, /*neg_b=*/false, bfrag,
          /*c_mod=*/(short)0, acc[nt], /*reuse_a=*/false, /*reuse_b=*/false);
    }
  }

  // ---- epilogue: D layout row=(lane/16)*8+r, col=lane%16 ----
  #pragma unroll
  for (int nt = 0; nt < 8; ++nt) {
    const int col = nbase + nt * 16 + (lane & 15);
    const float bv = bias[col];
    #pragma unroll
    for (int r = 0; r < 8; ++r) {
      const int row = m0 + mstrip * 16 + (lane >> 4) * 8 + r;
      currents[(size_t)row * F_ + col] = acc[nt][r] + bv;
    }
  }
}

// ---------------------------------------------------------------------------
// LIF scan + per-timestep LayerNorm. One block per batch row (sequential in T).
// thread f owns v_mem[f]; block reduction (wave32 shuffle + LDS) for mean.
// s in {0,1} => E[s^2] = E[s] => var = mu - mu^2 exactly.
// ---------------------------------------------------------------------------
__global__ __launch_bounds__(256)
void lif_scan_ln(const float* __restrict__ currents,
                 const float* __restrict__ gamma,
                 const float* __restrict__ beta,
                 float* __restrict__ out) {
  const int b    = blockIdx.x;
  const int f    = threadIdx.x;
  const int wave = f >> 5;
  const int lane = f & 31;

  __shared__ float partial[8];
  __shared__ float s_mu;

  const float alpha = __expf(-1.0f / 0.02f);   // exp(-1/tau_mem)
  const float oma   = 1.0f - alpha;

  float v = 0.0f;
  const float* curb = currents + (size_t)b * T_ * F_;
  float*       outb = out      + (size_t)b * T_ * F_;

  for (int t = 0; t < T_; ++t) {
    const float cur = curb[(size_t)t * F_ + f];
    v = alpha * v + oma * cur;
    const float s = (v - 0.5f > 0.0f) ? 1.0f : 0.0f;  // THRESH=0.5
    v = v * (1.0f - s);                               // RESET=0

    // block-wide sum of s: wave32 butterfly, then 8 partials via LDS
    float x = s;
    #pragma unroll
    for (int off = 16; off > 0; off >>= 1) x += __shfl_xor(x, off, 32);
    if (lane == 0) partial[wave] = x;
    __syncthreads();
    if (f == 0) {
      float tot = 0.0f;
      #pragma unroll
      for (int w = 0; w < 8; ++w) tot += partial[w];
      s_mu = tot * (1.0f / (float)F_);
    }
    __syncthreads();

    const float mu  = s_mu;
    const float var = mu - mu * mu;
    const float inv = __frsqrt_rn(var + 1e-6f);
    outb[(size_t)t * F_ + f] =
        (s - mu) * inv * gamma[(size_t)t * F_ + f] + beta[(size_t)t * F_ + f];
    __syncthreads();  // protect partial/s_mu before next iteration overwrites
  }
}

// ---------------------------------------------------------------------------
extern "C" void kernel_launch(void* const* d_in, const int* in_sizes, int n_in,
                              void* d_out, int out_size, void* d_ws, size_t ws_size,
                              hipStream_t stream) {
  const float* spikes = (const float*)d_in[0];  // [128,256,256]
  const float* W      = (const float*)d_in[1];  // [256,256]
  const float* bias   = (const float*)d_in[2];  // [256]
  const float* gamma  = (const float*)d_in[3];  // [256,256] (T,F)
  const float* beta   = (const float*)d_in[4];  // [256,256] (T,F)
  float* out      = (float*)d_out;              // [128,256,256]
  float* currents = (float*)d_ws;               // 32 MB scratch

  // GEMM: M = B*T = 32768 rows, 64-row tiles -> 512 blocks
  lif_gemm_wmma<<<(B_ * T_) / MTILE, 256, 0, stream>>>(spikes, W, bias, currents);
  // Scan: one block per batch row
  lif_scan_ln<<<B_, 256, 0, stream>>>(currents, gamma, beta, out);
}
